// Qwen3_5_Block_62612033241795
// MI455X (gfx1250) — matrix operand order repaired and checked by reading, not compiled, not run on hardware
//
#include <hip/hip_runtime.h>
#include <hip/hip_bf16.h>

#define H_    8
#define KVH_  2
#define D_    256
#define HID_  1024
#define INTER_ 3584
#define B_    4
#define S_    2048
#define NT_   (B_*S_)
#define ROT_  64
#define QGN_  4096      // H*D*2
#define HD_   2048      // H*D
#define KVD_  512       // KVH*D

typedef unsigned short u16;
typedef u16   su8   __attribute__((ext_vector_type(8)));
typedef u16   su16  __attribute__((ext_vector_type(16)));
typedef __bf16 v16bf __attribute__((ext_vector_type(16)));
typedef float v8f   __attribute__((ext_vector_type(8)));

__device__ __forceinline__ u16 f2b(float f) {
    unsigned u = __float_as_uint(f);
    unsigned r = 0x7FFFu + ((u >> 16) & 1u);
    return (u16)((u + r) >> 16);
}

__device__ __forceinline__ v16bf pack16(su8 lo, su8 hi) {
    su16 v;
#pragma unroll
    for (int i = 0; i < 8; ++i) { v[i] = lo[i]; v[i + 8] = hi[i]; }
    return __builtin_bit_cast(v16bf, v);
}

// A fragment: 16x32 (MxK), row-major source with leading dim lda (elements).
// ISA layout: lanes 0-15 row=lane, K={0..7,16..23}; lanes 16-31 row=lane-16, K={8..15,24..31}
__device__ __forceinline__ v16bf load_a_frag(const u16* __restrict__ A, int lda) {
    int lane = threadIdx.x & 31;
    int row  = lane & 15;
    int koff = (lane < 16) ? 0 : 8;
    const u16* p = A + (size_t)row * lda + koff;
    return pack16(*(const su8*)p, *(const su8*)(p + 16));
}

// B fragment: 32x16 (KxN) built from row-major W[N][K].
// ISA layout: lanes 0-15 col=lane K={0..7,16..23}; lanes 16-31 col=lane-16 K={8..15,24..31}
__device__ __forceinline__ v16bf load_b_frag(const u16* __restrict__ W, int ldw) {
    int lane = threadIdx.x & 31;
    int n  = lane & 15;
    int k0 = (lane < 16) ? 0 : 8;
    const u16* p = W + (size_t)n * ldw + k0;
    return pack16(*(const su8*)p, *(const su8*)(p + 16));
}

__device__ __forceinline__ v8f wmma_bf16(v16bf a, v16bf b, v8f c) {
    return __builtin_amdgcn_wmma_f32_16x16x32_bf16(false, a, false, b, (short)0, c, false, false);
}

__device__ __forceinline__ float silu(float x) { return x / (1.f + __expf(-x)); }

// CDNA5 async global->LDS copy (ASYNCcnt-tracked), 16 bytes per lane.
__device__ __forceinline__ void async_ld_b128(unsigned lds_addr, const u16* gaddr) {
    asm volatile("global_load_async_to_lds_b128 %0, %1, off"
                 :: "v"(lds_addr), "v"(gaddr) : "memory");
}
__device__ __forceinline__ void wait_async0() {
    asm volatile("s_wait_asynccnt 0x0" ::: "memory");
}

// ---------------- elementwise / norm kernels ----------------

__global__ void q35_f2b_kernel(const float* __restrict__ in, u16* __restrict__ out, size_t n) {
    size_t i = (size_t)blockIdx.x * blockDim.x + threadIdx.x;
    if (i < n) out[i] = f2b(in[i]);
}

__global__ __launch_bounds__(256)
void q35_rmsnorm_bf16_kernel(const float* __restrict__ x, const float* __restrict__ w,
                             u16* __restrict__ out, int cols) {
    const int t = blockIdx.x;
    const float* row = x + (size_t)t * cols;
    float ss = 0.f;
    for (int c = threadIdx.x; c < cols; c += 256) { float v = row[c]; ss += v * v; }
#pragma unroll
    for (int m = 16; m; m >>= 1) ss += __shfl_xor(ss, m, 32);
    __shared__ float red[8];
    if ((threadIdx.x & 31) == 0) red[threadIdx.x >> 5] = ss;
    __syncthreads();
    float tot = 0.f;
#pragma unroll
    for (int i = 0; i < 8; ++i) tot += red[i];
    float rinv = rsqrtf(tot / (float)cols + 1e-6f);
    for (int c = threadIdx.x; c < cols; c += 256)
        out[(size_t)t * cols + c] = f2b(row[c] * rinv * w[c]);
}

// per-(token, head) rmsnorm over D + partial RoPE; heads 0..H-1 = q, H..H+KVH-1 = k
__global__ __launch_bounds__(256)
void q35_qk_norm_rope_kernel(const float* __restrict__ qg, const float* __restrict__ kbuf,
                             const float* __restrict__ qn_w, const float* __restrict__ kn_w,
                             const int* __restrict__ pos_ids,
                             u16* __restrict__ Qb, u16* __restrict__ Kb) {
    const int tok  = blockIdx.x;
    const int head = blockIdx.y;
    const int d    = threadIdx.x;
    const bool isq = head < H_;
    const float* row = isq ? (qg + (size_t)tok * QGN_ + head * D_)
                           : (kbuf + (size_t)tok * KVD_ + (head - H_) * D_);
    float v = row[d];
    float ss = v * v;
#pragma unroll
    for (int m = 16; m; m >>= 1) ss += __shfl_xor(ss, m, 32);
    __shared__ float red[8];
    if ((threadIdx.x & 31) == 0) red[threadIdx.x >> 5] = ss;
    __syncthreads();
    float tot = 0.f;
#pragma unroll
    for (int i = 0; i < 8; ++i) tot += red[i];
    float rinv = rsqrtf(tot / (float)D_ + 1e-6f);
    const float* wv = isq ? qn_w : kn_w;
    float y = v * rinv * wv[d];
    float outv = y;
    const int s = tok & (S_ - 1);
    if (d < ROT_) {
        float p = (float)pos_ids[s];
        int j = d & 31;                                  // freq index
        float inv_freq = __powf(1.0e7f, -(float)j * (1.0f / 32.0f));
        float ang = p * inv_freq;
        float sv, cv;
        __sincosf(ang, &sv, &cv);
        int pd = (d < 32) ? d + 32 : d - 32;
        float py = row[pd] * rinv * wv[pd];
        outv = (d < 32) ? (y * cv - py * sv) : (y * cv + py * sv);
    }
    const int b = tok >> 11;                             // / S_
    if (isq) Qb[((size_t)(b * H_ + head) * S_ + s) * D_ + d] = f2b(outv);
    else     Kb[((size_t)(b * KVH_ + (head - H_)) * S_ + s) * D_ + d] = f2b(outv);
}

// V [token][kvh*D] fp32 -> Vt [b][kvh][d][s] bf16 (transposed so P@V B-fragments are contiguous)
__global__ void q35_vt_kernel(const float* __restrict__ vbuf, u16* __restrict__ Vt) {
    size_t idx = (size_t)blockIdx.x * blockDim.x + threadIdx.x;
    if (idx >= (size_t)NT_ * KVD_) return;
    int s = (int)(idx & (S_ - 1));
    size_t r = idx >> 11;
    int d = (int)(r & (D_ - 1));
    r >>= 8;
    int kvh = (int)(r & (KVH_ - 1));
    int b = (int)(r >> 1);
    Vt[idx] = f2b(vbuf[((size_t)(b * S_ + s)) * KVD_ + kvh * D_ + d]);
}

// ---------------- WMMA GEMM with async-LDS double buffering ----------------
// C[M,N] = A[M,K] (bf16) x W[N,K]^T (bf16)
// Workgroup: 256 threads = 8 waves (2 M x 4 N), tile 64x256, K step 32.
// mode 0: outf = acc ; mode 1: outf = acc + aux ; mode 2: outb = bf16(silu(aux)*acc)
#define GA_ROWS 64
#define GB_ROWS 256
#define GK      32
#define TILE_U16 ((GA_ROWS + GB_ROWS) * GK)   // 10240 u16 = 20 KB

__global__ __launch_bounds__(256)
void q35_gemm_wmma_kernel(const u16* __restrict__ A, const u16* __restrict__ W,
                          float* __restrict__ outf, u16* __restrict__ outb,
                          const float* __restrict__ aux,
                          int M, int N, int K, int mode) {
    const int n0 = blockIdx.x * GB_ROWS;
    const int m0 = blockIdx.y * GA_ROWS;
    const int tid = threadIdx.x;
    const int wid = tid >> 5;
    const int wm = wid & 1;        // 0..1  -> 32-row slab
    const int wn = wid >> 1;       // 0..3  -> 64-col slab

    __shared__ u16 smem[2 * TILE_U16];  // 40 KB double buffer

    const int lr = tid >> 2;            // 0..63
    const int lc = (tid & 3) * 8;       // 0,8,16,24 (u16), 16B chunks

    auto stage = [&](int kk, int buf) {
        u16* Ab = smem + buf * TILE_U16;
        u16* Bb = Ab + GA_ROWS * GK;
        async_ld_b128((unsigned)(uintptr_t)(Ab + lr * GK + lc),
                      A + (size_t)(m0 + lr) * K + kk + lc);
#pragma unroll
        for (int j = 0; j < 4; ++j) {
            int nn = lr + 64 * j;
            async_ld_b128((unsigned)(uintptr_t)(Bb + nn * GK + lc),
                          W + (size_t)(n0 + nn) * K + kk + lc);
        }
    };

    stage(0, 0);

    v8f acc[2][4] = {};
    const int nsteps = K >> 5;
    for (int s = 0; s < nsteps; ++s) {
        wait_async0();          // my async stores into LDS complete
        __syncthreads();        // everyone's complete -> current buffer valid
        if (s + 1 < nsteps) stage((s + 1) << 5, (s + 1) & 1);  // overlap next tile
        const u16* Ab = smem + (s & 1) * TILE_U16;
        const u16* Bb = Ab + GA_ROWS * GK;
        v16bf a0 = load_a_frag(Ab + (wm * 32) * GK, GK);
        v16bf a1 = load_a_frag(Ab + (wm * 32 + 16) * GK, GK);
#pragma unroll
        for (int t = 0; t < 4; ++t) {
            v16bf b = load_b_frag(Bb + (wn * 64 + 16 * t) * GK, GK);
            acc[0][t] = wmma_bf16(a0, b, acc[0][t]);
            acc[1][t] = wmma_bf16(a1, b, acc[1][t]);
        }
    }

    const int lane = tid & 31;
    const int col = lane & 15;
    const int rb = (lane < 16) ? 0 : 8;
#pragma unroll
    for (int mt = 0; mt < 2; ++mt) {
#pragma unroll
        for (int t = 0; t < 4; ++t) {
#pragma unroll
            for (int i = 0; i < 8; ++i) {
                int row = m0 + wm * 32 + mt * 16 + rb + i;
                int cl  = n0 + wn * 64 + 16 * t + col;
                size_t idx = (size_t)row * N + cl;
                float v = acc[mt][t][i];
                if (mode == 1)      outf[idx] = v + aux[idx];
                else if (mode == 2) outb[idx] = f2b(silu(aux[idx]) * v);
                else                outf[idx] = v;
            }
        }
    }
}

// ---------------- flash attention (non-causal), fused silu(gate) epilogue ----------------
// grid: (S/16, H, B), block: 32 (one wave). Output: aob [token][H*D] bf16.
__global__ __launch_bounds__(32)
void q35_attn_wmma_kernel(const u16* __restrict__ Qb, const u16* __restrict__ Kb,
                          const u16* __restrict__ Vt, const float* __restrict__ qg,
                          u16* __restrict__ aob) {
    const int qb = blockIdx.x, h = blockIdx.y, b = blockIdx.z;
    const int kvh = h / (H_ / KVH_);
    const int lane = threadIdx.x & 31;
    const int col = lane & 15;
    const int rb = (lane < 16) ? 0 : 8;
    const u16* Q  = Qb + ((size_t)(b * H_ + h) * S_ + qb * 16) * D_;
    const u16* Kp = Kb + (size_t)(b * KVH_ + kvh) * S_ * D_;
    const u16* Vp = Vt + (size_t)(b * KVH_ + kvh) * D_ * S_;

    v16bf qf[8];
#pragma unroll
    for (int i = 0; i < 8; ++i) qf[i] = load_a_frag(Q + 32 * i, D_);

    v8f o[16] = {};
    float mi[8], li[8];
#pragma unroll
    for (int i = 0; i < 8; ++i) { mi[i] = -1e30f; li[i] = 0.f; }

    __shared__ u16 Pl[16 * 32];

    for (int kb = 0; kb < S_; kb += 32) {
        v8f s0 = {}, s1 = {};
#pragma unroll
        for (int d = 0; d < D_; d += 32) {
            v16bf k0 = load_b_frag(Kp + (size_t)kb * D_ + d, D_);
            v16bf k1 = load_b_frag(Kp + (size_t)(kb + 16) * D_ + d, D_);
            s0 = wmma_bf16(qf[d >> 5], k0, s0);
            s1 = wmma_bf16(qf[d >> 5], k1, s1);
        }
        float corr[8];
#pragma unroll
        for (int i = 0; i < 8; ++i) {
            float a  = s0[i] * 0.0625f;     // 1/sqrt(256)
            float bb = s1[i] * 0.0625f;
            float mx = fmaxf(a, bb);
#pragma unroll
            for (int msk = 8; msk; msk >>= 1) mx = fmaxf(mx, __shfl_xor(mx, msk, 32));
            float mnew = fmaxf(mi[i], mx);
            float c  = __expf(mi[i] - mnew);
            float p0 = __expf(a - mnew);
            float p1 = __expf(bb - mnew);
            float rs = p0 + p1;
#pragma unroll
            for (int msk = 8; msk; msk >>= 1) rs += __shfl_xor(rs, msk, 32);
            li[i] = li[i] * c + rs;
            mi[i] = mnew;
            corr[i] = c;
            Pl[(rb + i) * 32 + col]      = f2b(p0);
            Pl[(rb + i) * 32 + 16 + col] = f2b(p1);
        }
#pragma unroll
        for (int t = 0; t < 16; ++t)
#pragma unroll
            for (int i = 0; i < 8; ++i) o[t][i] *= corr[i];
        __syncthreads();
        {
            // re-read P (16x32) from LDS in A-fragment layout
            int row = lane & 15;
            int koff = (lane < 16) ? 0 : 8;
            const u16* p = &Pl[row * 32 + koff];
            v16bf pa = pack16(*(const su8*)p, *(const su8*)(p + 16));
#pragma unroll
            for (int t = 0; t < 16; ++t) {
                v16bf bv = load_b_frag(Vp + (size_t)(16 * t) * S_ + kb, S_);
                o[t] = wmma_bf16(pa, bv, o[t]);
            }
        }
        __syncthreads();
    }

    // epilogue: normalize, fuse silu(gate), store bf16
#pragma unroll
    for (int t = 0; t < 16; ++t) {
#pragma unroll
        for (int i = 0; i < 8; ++i) {
            int s = qb * 16 + rb + i;
            size_t tok = (size_t)b * S_ + s;
            int c = h * D_ + 16 * t + col;
            float gt = qg[tok * QGN_ + HD_ + c];
            float val = (o[t][i] / li[i]) * silu(gt);
            aob[tok * (size_t)HD_ + c] = f2b(val);
        }
    }
}

// ---------------- host ----------------

extern "C" void kernel_launch(void* const* d_in, const int* in_sizes, int n_in,
                              void* d_out, int out_size, void* d_ws, size_t ws_size,
                              hipStream_t stream) {
    const float* x    = (const float*)d_in[0];
    const int*   pos  = (const int*)d_in[1];
    const float* ln1w = (const float*)d_in[2];
    const float* ln2w = (const float*)d_in[3];
    const float* Wq   = (const float*)d_in[4];
    const float* Wk   = (const float*)d_in[5];
    const float* Wv   = (const float*)d_in[6];
    const float* Wo   = (const float*)d_in[7];
    const float* qnw  = (const float*)d_in[8];
    const float* knw  = (const float*)d_in[9];
    const float* Wg   = (const float*)d_in[10];
    const float* Wu   = (const float*)d_in[11];
    const float* Wd   = (const float*)d_in[12];
    float* out = (float*)d_out;

    char* ws = (char*)d_ws;
    size_t off = 0;
    auto alloc = [&](size_t bytes) -> char* {
        char* p = ws + off;
        off += (bytes + 255) & ~(size_t)255;
        return p;
    };

    u16*   hb   = (u16*)alloc((size_t)NT_ * HID_ * 2);
    u16*   wqb  = (u16*)alloc((size_t)QGN_ * HID_ * 2);
    u16*   wkb  = (u16*)alloc((size_t)KVD_ * HID_ * 2);
    u16*   wvb  = (u16*)alloc((size_t)KVD_ * HID_ * 2);
    u16*   wob  = (u16*)alloc((size_t)HID_ * HD_ * 2);
    u16*   wgb  = (u16*)alloc((size_t)INTER_ * HID_ * 2);
    u16*   wub  = (u16*)alloc((size_t)INTER_ * HID_ * 2);
    u16*   wdb  = (u16*)alloc((size_t)HID_ * INTER_ * 2);
    float* qg   = (float*)alloc((size_t)NT_ * QGN_ * 4);   // later reused as MLP gate buffer
    float* kbuf = (float*)alloc((size_t)NT_ * KVD_ * 4);
    float* vbuf = (float*)alloc((size_t)NT_ * KVD_ * 4);
    u16*   Qb   = (u16*)alloc((size_t)NT_ * HD_ * 2);
    u16*   Kb   = (u16*)alloc((size_t)NT_ * KVD_ * 2);
    u16*   Vt   = (u16*)alloc((size_t)NT_ * KVD_ * 2);
    u16*   aob  = (u16*)alloc((size_t)NT_ * HD_ * 2);
    float* x2   = (float*)alloc((size_t)NT_ * HID_ * 4);
    u16*   h2b  = (u16*)alloc((size_t)NT_ * HID_ * 2);
    u16*   gub  = (u16*)alloc((size_t)NT_ * INTER_ * 2);
    float* gbuf = qg;  // reuse: qg dead after attention; MLP gate (NT x INTER) fits inside

    auto cvt = [&](const float* src, u16* dst, size_t n) {
        q35_f2b_kernel<<<(unsigned)((n + 255) / 256), 256, 0, stream>>>(src, dst, n);
    };

    // 1) convert weights to bf16
    cvt(Wq, wqb, (size_t)QGN_ * HID_);
    cvt(Wk, wkb, (size_t)KVD_ * HID_);
    cvt(Wv, wvb, (size_t)KVD_ * HID_);
    cvt(Wo, wob, (size_t)HID_ * HD_);
    cvt(Wg, wgb, (size_t)INTER_ * HID_);
    cvt(Wu, wub, (size_t)INTER_ * HID_);
    cvt(Wd, wdb, (size_t)HID_ * INTER_);

    // 2) h = rmsnorm(x, ln1_w) -> bf16
    q35_rmsnorm_bf16_kernel<<<NT_, 256, 0, stream>>>(x, ln1w, hb, HID_);

    // 3) qg = h @ Wq^T ; k = h @ Wk^T ; v = h @ Wv^T
    q35_gemm_wmma_kernel<<<dim3(QGN_ / 256, NT_ / 64), 256, 0, stream>>>(
        hb, wqb, qg, nullptr, nullptr, NT_, QGN_, HID_, 0);
    q35_gemm_wmma_kernel<<<dim3(KVD_ / 256, NT_ / 64), 256, 0, stream>>>(
        hb, wkb, kbuf, nullptr, nullptr, NT_, KVD_, HID_, 0);
    q35_gemm_wmma_kernel<<<dim3(KVD_ / 256, NT_ / 64), 256, 0, stream>>>(
        hb, wvb, vbuf, nullptr, nullptr, NT_, KVD_, HID_, 0);

    // 4) q/k head-norm + RoPE ; transpose V
    q35_qk_norm_rope_kernel<<<dim3(NT_, H_ + KVH_), 256, 0, stream>>>(
        qg, kbuf, qnw, knw, pos, Qb, Kb);
    q35_vt_kernel<<<(unsigned)(((size_t)NT_ * KVD_ + 255) / 256), 256, 0, stream>>>(vbuf, Vt);

    // 5) attention with fused silu(gate) -> aob bf16
    q35_attn_wmma_kernel<<<dim3(S_ / 16, H_, B_), 32, 0, stream>>>(Qb, Kb, Vt, qg, aob);

    // 6) x2 = x + aob @ Wo^T
    q35_gemm_wmma_kernel<<<dim3(HID_ / 256, NT_ / 64), 256, 0, stream>>>(
        aob, wob, x2, nullptr, x, NT_, HID_, HD_, 1);

    // 7) h2 = rmsnorm(x2, ln2_w) -> bf16
    q35_rmsnorm_bf16_kernel<<<NT_, 256, 0, stream>>>(x2, ln2w, h2b, HID_);

    // 8) g = h2 @ Wg^T ; gu = silu(g) * (h2 @ Wu^T) -> bf16
    q35_gemm_wmma_kernel<<<dim3(INTER_ / 256, NT_ / 64), 256, 0, stream>>>(
        h2b, wgb, gbuf, nullptr, nullptr, NT_, INTER_, HID_, 0);
    q35_gemm_wmma_kernel<<<dim3(INTER_ / 256, NT_ / 64), 256, 0, stream>>>(
        h2b, wub, nullptr, gub, gbuf, NT_, INTER_, HID_, 2);

    // 9) out = x2 + gu @ Wd^T
    q35_gemm_wmma_kernel<<<dim3(HID_ / 256, NT_ / 64), 256, 0, stream>>>(
        gub, wdb, out, nullptr, x2, NT_, HID_, INTER_, 1);

    (void)in_sizes; (void)n_in; (void)out_size; (void)ws_size;
}
